// EquivariantCorrectionHead_39711267619357
// MI455X (gfx1250) — matrix operand order repaired and checked by reading, not compiled, not run on hardware
//
#include <hip/hip_runtime.h>
#include <cmath>
#include <complex>
#include <vector>
#include <algorithm>

typedef __attribute__((ext_vector_type(16))) __bf16 v16bf;
typedef __attribute__((ext_vector_type(8)))  float  v8f;

struct C222Arg { float c[125]; };

#define WAVES   4
#define THREADS (WAVES * 32)

__device__ __forceinline__ v8f wmma_bf16(v16bf a, v16bf b, v8f c) {
  // D = A(16x32 bf16) x B(32x16 bf16) + C(16x16 f32)
  return __builtin_amdgcn_wmma_f32_16x16x32_bf16(false, a, false, b, (short)0, c,
                                                 false, false);
}

__global__ __launch_bounds__(THREADS, 1)
void eqhead_kernel(const float* __restrict__ g_s,    const float* __restrict__ g_t2s,
                   const float* __restrict__ g_w000, const float* __restrict__ g_w110,
                   const float* __restrict__ g_w011, const float* __restrict__ g_w101,
                   const float* __restrict__ g_w111, const float* __restrict__ g_v010,
                   const float* __restrict__ g_v100, const float* __restrict__ g_v110,
                   float* __restrict__ g_out, int Btot, C222Arg cc)
{
  // ---- weights staged in LDS as pre-scaled bf16, laid out for B-fragments ----
  __shared__ __bf16 sw000[256 * 32];   // [u*16+v][w], scaled c0
  __shared__ __bf16 sw110[96 * 32];    // [u*9+v pad96][w], scaled c0/sqrt5
  __shared__ __bf16 sw011[32 * 288];   // [u pad32][v*32+w], scaled c2/sqrt5
  __shared__ __bf16 sw101[32 * 288];   // [v pad32][u*32+w], scaled c2/sqrt5
  __shared__ __bf16 sw111[96 * 32];    // [u*9+v pad96][w], scaled c2
  __shared__ __bf16 sv010[32 * 32];    // [u][v], scaled d/sqrt5
  __shared__ __bf16 sv100t[32 * 32];   // [v][u] (= v100^T), scaled d/sqrt5
  __shared__ __bf16 sv110t[32 * 32];   // [v][u] (= v110^T), unscaled
  __shared__ float  sC[125];           // Wigner 3j(2,2,2), real basis
  // ---- per-wave (16-row tile) staging ----
  __shared__ float  sS [WAVES][16][16];
  __shared__ float  sT [WAVES][16][9][8];   // inner padded 5->8 (32B rows)
  __shared__ float  sH0[WAVES][16][32];
  __shared__ float  sH2[WAVES][16][32][5];
  __shared__ float  sA2[WAVES][16][32][5];
  __shared__ float  sR [WAVES][16][32];
  __shared__ float  sM [WAVES][16][32];

  const int tid = threadIdx.x;
  const float inv5 = 0.44721359549995794f;            // 1/sqrt(5)
  const float c0   = 1.0f / sqrtf(337.0f);            // 1/sqrt(S*S+81)
  const float c2   = sqrtf(5.0f / 369.0f);            // sqrt(5/(18S+81))
  const float dsc  = sqrtf(5.0f / 3072.0f);           // sqrt(5/(3H^2))
  const float c0i5 = c0 * inv5, c2i5 = c2 * inv5, di5 = dsc * inv5;

  for (int i = tid; i < 256 * 32; i += THREADS) sw000[i] = (__bf16)(c0 * g_w000[i]);
  for (int i = tid; i < 96 * 32; i += THREADS) {
    int r = i >> 5, w = i & 31;
    sw110[i] = (__bf16)((r < 81) ? c0i5 * g_w110[r * 32 + w] : 0.0f);
  }
  for (int i = tid; i < 32 * 288; i += THREADS) {
    int r = i / 288, c = i - r * 288;
    sw011[i] = (__bf16)((r < 16) ? c2i5 * g_w011[r * 288 + c] : 0.0f);
  }
  for (int i = tid; i < 32 * 288; i += THREADS) {
    int v = i / 288, c = i - v * 288, u = c >> 5, w = c & 31;
    sw101[i] = (__bf16)((v < 16) ? c2i5 * g_w101[u * 512 + v * 32 + w] : 0.0f);
  }
  for (int i = tid; i < 96 * 32; i += THREADS) {
    int r = i >> 5, w = i & 31;
    sw111[i] = (__bf16)((r < 81) ? c2 * g_w111[r * 32 + w] : 0.0f);
  }
  for (int i = tid; i < 1024; i += THREADS) {
    int r = i >> 5, c = i & 31;
    sv010[i]  = (__bf16)(di5 * g_v010[i]);
    sv100t[i] = (__bf16)(di5 * g_v100[c * 32 + r]);
    sv110t[i] = (__bf16)(g_v110[c * 32 + r]);
  }
  for (int i = tid; i < 125; i += THREADS) sC[i] = cc.c[i];
  __syncthreads();

  const int wv = tid >> 5, lane = tid & 31, lm = lane & 15, lg = lane >> 4;
  const int b0 = (blockIdx.x * WAVES + wv) * 16;
  if (b0 >= Btot) return;

  // A-fragment K index for (lane,half): contiguous k = 16*lg + h (matches B fills)
  auto AK = [lg](int h) {
    return h + (lg << 4);
  };

  // ---- load s, build t = [selected 8 kernels, kernel-sum] ----
  for (int idx = lane; idx < 256; idx += 32)
    sS[wv][idx >> 4][idx & 15] = g_s[(size_t)b0 * 16 + idx];
  {
    const unsigned long long selmask =
        (1ull << 0) | (1ull << 1) | (1ull << 2) | (1ull << 4) |
        (1ull << 24) | (1ull << 26) | (1ull << 35) | (1ull << 38);
    const float* base = g_t2s + (size_t)(b0 + lm) * 200 + lg * 100;
    float ps[5] = {0.f, 0.f, 0.f, 0.f, 0.f};
    for (int kk = 0; kk < 20; ++kk) {
      float v[5];
#pragma unroll
      for (int i = 0; i < 5; ++i) { v[i] = base[kk * 5 + i]; ps[i] += v[i]; }
      int gk = lg * 20 + kk;
      if ((selmask >> gk) & 1ull) {
        int j = __popcll(selmask & ((1ull << gk) - 1ull));
#pragma unroll
        for (int i = 0; i < 5; ++i) sT[wv][lm][j][i] = v[i];
      }
    }
#pragma unroll
    for (int i = 0; i < 5; ++i) ps[i] += __shfl_xor(ps[i], 16);
    if (lg == 0)
#pragma unroll
      for (int i = 0; i < 5; ++i) sT[wv][lm][8][i] = ps[i];
  }

  // C222 in registers (fully constant-indexed after unrolling)
  float cr[125];
#pragma unroll
  for (int i = 0; i < 125; ++i) cr[i] = sC[i];

  // ================= h0 = outer(s,s)@w000 + (t.t)@w110  (scales folded) ======
  {
    v8f acc0 = {}, acc1 = {};
    for (int ch = 0; ch < 8; ++ch) {           // K = 256 in 8 chunks
      v16bf a;
#pragma unroll
      for (int h = 0; h < 16; ++h) {
        int col = ch * 32 + AK(h);
        a[h] = (__bf16)(sS[wv][lm][col >> 4] * sS[wv][lm][col & 15]);
      }
      v16bf b0f, b1f;
#pragma unroll
      for (int h = 0; h < 16; ++h) {
        int k = ch * 32 + lg * 16 + h;
        b0f[h] = sw000[k * 32 + lm];
        b1f[h] = sw000[k * 32 + 16 + lm];
      }
      acc0 = wmma_bf16(a, b0f, acc0);
      acc1 = wmma_bf16(a, b1f, acc1);
    }
    for (int ch = 0; ch < 3; ++ch) {           // K = 81 padded to 96
      v16bf a;
#pragma unroll
      for (int h = 0; h < 16; ++h) {
        int col = ch * 32 + AK(h);
        float val = 0.0f;
        if (col < 81) {
          int u = col / 9, vv = col - u * 9;
#pragma unroll
          for (int i = 0; i < 5; ++i) val += sT[wv][lm][u][i] * sT[wv][lm][vv][i];
        }
        a[h] = (__bf16)val;
      }
      v16bf b0f, b1f;
#pragma unroll
      for (int h = 0; h < 16; ++h) {
        int k = ch * 32 + lg * 16 + h;
        b0f[h] = sw110[k * 32 + lm];
        b1f[h] = sw110[k * 32 + 16 + lm];
      }
      acc0 = wmma_bf16(a, b0f, acc0);
      acc1 = wmma_bf16(a, b1f, acc1);
    }
#pragma unroll
    for (int r = 0; r < 8; ++r) {
      int m = r + 8 * lg;
      sH0[wv][m][lm] = acc0[r];
      sH0[wv][m][16 + lm] = acc1[r];
    }
  }

  // ===== h2 accumulators live in registers; lane owns (rows r+8*lg, w in {lm, lm+16})
  v8f h2acc[5][2];
#pragma unroll
  for (int k2 = 0; k2 < 5; ++k2) {
    v8f z = {};
    h2acc[k2][0] = z;
    h2acc[k2][1] = z;
  }

  // ================= h2 base: u2 @ w111, all 5 k at once =================
  for (int ch = 0; ch < 3; ++ch) {
    v16bf af[5];
#pragma unroll
    for (int h = 0; h < 16; ++h) {
      int col = ch * 32 + AK(h);
      float val[5] = {0.f, 0.f, 0.f, 0.f, 0.f};
      if (col < 81) {
        int u = col / 9, vv = col - u * 9;
        float tu[5], tv[5];
#pragma unroll
        for (int i = 0; i < 5; ++i) {
          tu[i] = sT[wv][lm][u][i];
          tv[i] = sT[wv][lm][vv][i];
        }
#pragma unroll
        for (int i = 0; i < 5; ++i)
#pragma unroll
          for (int j = 0; j < 5; ++j) {
            float p = tu[i] * tv[j];
#pragma unroll
            for (int k2 = 0; k2 < 5; ++k2) val[k2] += p * cr[i * 25 + j * 5 + k2];
          }
      }
#pragma unroll
      for (int k2 = 0; k2 < 5; ++k2) af[k2][h] = (__bf16)val[k2];
    }
    v16bf b0f, b1f;
#pragma unroll
    for (int h = 0; h < 16; ++h) {
      int k = ch * 32 + lg * 16 + h;
      b0f[h] = sw111[k * 32 + lm];
      b1f[h] = sw111[k * 32 + 16 + lm];
    }
#pragma unroll
    for (int k2 = 0; k2 < 5; ++k2) {
      h2acc[k2][0] = wmma_bf16(af[k2], b0f, h2acc[k2][0]);
      h2acc[k2][1] = wmma_bf16(af[k2], b1f, h2acc[k2][1]);
    }
  }

  // ======== 011 + 101 paths, merged: h2 += (P011 + P101) * t  (reg accum) =====
  {
    v16bf afs;   // A = s, K=16 zero-padded to 32
#pragma unroll
    for (int h = 0; h < 16; ++h) {
      int k = AK(h);
      afs[h] = (k < 16) ? (__bf16)sS[wv][lm][k] : (__bf16)0.0f;
    }
    for (int q = 0; q < 9; ++q) {       // q = v for 011, q = u for 101
      float tq[8][5];
#pragma unroll
      for (int r = 0; r < 8; ++r)
#pragma unroll
        for (int k2 = 0; k2 < 5; ++k2) tq[r][k2] = sT[wv][r + 8 * lg][q][k2];
#pragma unroll
      for (int wh = 0; wh < 2; ++wh) {
        int nt = q * 2 + wh;
        v16bf bA, bB;
#pragma unroll
        for (int h = 0; h < 16; ++h) {
          int off = (lg * 16 + h) * 288 + nt * 16 + lm;
          bA[h] = sw011[off];
          bB[h] = sw101[off];
        }
        v8f zA = {}, zB = {};
        v8f dA = wmma_bf16(afs, bA, zA);
        v8f dB = wmma_bf16(afs, bB, zB);
#pragma unroll
        for (int r = 0; r < 8; ++r) {
          float pv = dA[r] + dB[r];
#pragma unroll
          for (int k2 = 0; k2 < 5; ++k2) h2acc[k2][wh][r] += pv * tq[r][k2];
        }
      }
    }
  }

  // single h2 writeback to LDS (needed cross-lane for tp2 + finish)
#pragma unroll
  for (int wh = 0; wh < 2; ++wh)
#pragma unroll
    for (int r = 0; r < 8; ++r) {
      int m = r + 8 * lg, w = wh * 16 + lm;
#pragma unroll
      for (int k2 = 0; k2 < 5; ++k2) sH2[wv][m][w][k2] = h2acc[k2][wh][r];
    }

  // ================= tp2 GEMMs: a = h2 @ v110^T, r = h0 @ v010, m = h0 @ v100^T
  for (int k2 = 0; k2 < 5; ++k2) {
    v16bf a;
#pragma unroll
    for (int h = 0; h < 16; ++h) a[h] = (__bf16)sH2[wv][lm][AK(h)][k2];
    for (int nt = 0; nt < 2; ++nt) {
      v16bf b;
#pragma unroll
      for (int h = 0; h < 16; ++h) b[h] = sv110t[(lg * 16 + h) * 32 + nt * 16 + lm];
      v8f z = {};
      v8f dfr = wmma_bf16(a, b, z);
#pragma unroll
      for (int r = 0; r < 8; ++r) sA2[wv][r + 8 * lg][nt * 16 + lm][k2] = dfr[r];
    }
  }
  {
    v16bf ah0;
#pragma unroll
    for (int h = 0; h < 16; ++h) ah0[h] = (__bf16)sH0[wv][lm][AK(h)];
    for (int nt = 0; nt < 2; ++nt) {
      v16bf b1, b2f;
#pragma unroll
      for (int h = 0; h < 16; ++h) {
        b1[h]  = sv010[(lg * 16 + h) * 32 + nt * 16 + lm];
        b2f[h] = sv100t[(lg * 16 + h) * 32 + nt * 16 + lm];
      }
      v8f z0 = {}, z1 = {};
      v8f dr = wmma_bf16(ah0, b1, z0);
      v8f dm = wmma_bf16(ah0, b2f, z1);
#pragma unroll
      for (int r = 0; r < 8; ++r) {
        sR[wv][r + 8 * lg][nt * 16 + lm] = dr[r];
        sM[wv][r + 8 * lg][nt * 16 + lm] = dm[r];
      }
    }
  }

  // ================= finish: b2 = h2^T a ; out = r.h2 + m.h2 + d * b2:C ======
  if (lg == 0 && (b0 + lm) < Btot) {
    const int m = lm;
    float b2[5][5];
#pragma unroll
    for (int i = 0; i < 5; ++i)
#pragma unroll
      for (int j = 0; j < 5; ++j) {
        float acc = 0.0f;
        for (int u = 0; u < 32; ++u) acc += sH2[wv][m][u][i] * sA2[wv][m][u][j];
        b2[i][j] = acc;
      }
    float o[5];
#pragma unroll
    for (int k = 0; k < 5; ++k) {
      float acc = 0.0f;
      for (int u = 0; u < 32; ++u)
        acc += (sR[wv][m][u] + sM[wv][m][u]) * sH2[wv][m][u][k];
#pragma unroll
      for (int i = 0; i < 5; ++i)
#pragma unroll
        for (int j = 0; j < 5; ++j) acc += dsc * b2[i][j] * cr[i * 25 + j * 5 + k];
      o[k] = acc;
    }
#pragma unroll
    for (int k = 0; k < 5; ++k) g_out[(size_t)(b0 + m) * 5 + k] = o[k];
  }
}

// ===================== host: Wigner 3j(2,2,2) in real SH basis ================
static void compute_c222(float* outc) {
  typedef std::complex<double> cd;
  cd Jp[5][5], Jm[5][5], J[3][5][5], Q[5][5];
  for (int i = 0; i < 5; ++i)
    for (int j = 0; j < 5; ++j) {
      Jp[i][j] = Jm[i][j] = Q[i][j] = cd(0, 0);
      J[0][i][j] = J[1][i][j] = J[2][i][j] = cd(0, 0);
    }
  for (int i = 0; i < 4; ++i) {
    double m = i - 2.0;
    Jp[i + 1][i] = cd(std::sqrt(6.0 - m * (m + 1.0)), 0.0);
  }
  for (int i = 0; i < 5; ++i)
    for (int j = 0; j < 5; ++j) Jm[i][j] = std::conj(Jp[j][i]);
  for (int i = 0; i < 5; ++i)
    for (int j = 0; j < 5; ++j) {
      J[0][i][j] = (Jp[i][j] + Jm[i][j]) * 0.5;
      J[1][i][j] = (Jp[i][j] - Jm[i][j]) / cd(0.0, 2.0);
    }
  for (int i = 0; i < 5; ++i) J[2][i][i] = cd(i - 2.0, 0.0);
  const double s2 = std::sqrt(0.5);
  for (int m = -2; m <= 2; ++m) {
    int i = m + 2;
    double pm = (m & 1) ? -1.0 : 1.0;
    if (m < 0) { Q[i][i] = cd(0.0, s2); Q[i][2 - m] = cd(0.0, -pm * s2); }
    else if (m == 0) Q[i][i] = cd(1.0, 0.0);
    else { Q[i][2 - m] = cd(s2, 0.0); Q[i][i] = cd(pm * s2, 0.0); }
  }
  double G[3][5][5];
  for (int a = 0; a < 3; ++a) {
    cd T1[5][5], T2[5][5];
    for (int i = 0; i < 5; ++i)
      for (int j = 0; j < 5; ++j) {
        cd acc(0, 0);
        for (int k = 0; k < 5; ++k) acc += J[a][i][k] * std::conj(Q[j][k]);
        T1[i][j] = acc;
      }
    for (int i = 0; i < 5; ++i)
      for (int j = 0; j < 5; ++j) {
        cd acc(0, 0);
        for (int k = 0; k < 5; ++k) acc += Q[i][k] * T1[k][j];
        T2[i][j] = acc;
      }
    for (int i = 0; i < 5; ++i)
      for (int j = 0; j < 5; ++j) G[a][i][j] = T2[i][j].imag();  // Re(-i z) = Im z
  }
  std::vector<double> A(125 * 125, 0.0), M(125 * 125);
  for (int a = 0; a < 3; ++a) {
    std::fill(M.begin(), M.end(), 0.0);
    for (int p = 0; p < 125; ++p) {
      int i = p / 25, j = (p / 5) % 5, k = p % 5;
      for (int t = 0; t < 5; ++t) {
        M[p * 125 + t * 25 + j * 5 + k] += G[a][i][t];
        M[p * 125 + i * 25 + t * 5 + k] += G[a][j][t];
        M[p * 125 + i * 25 + j * 5 + t] += G[a][k][t];
      }
    }
    for (int p = 0; p < 125; ++p) {
      const double* mr = &M[p * 125];
      for (int q1 = 0; q1 < 125; ++q1) {
        double mv = mr[q1];
        if (mv == 0.0) continue;
        double* ar = &A[q1 * 125];
        for (int q2 = 0; q2 < 125; ++q2) ar[q2] += mv * mr[q2];
      }
    }
  }
  for (int d = 0; d < 125; ++d) A[d * 125 + d] += 1e-10;
  int piv[125];
  for (int c = 0; c < 125; ++c) {
    int p = c;
    double best = std::fabs(A[c * 125 + c]);
    for (int r = c + 1; r < 125; ++r) {
      double v = std::fabs(A[r * 125 + c]);
      if (v > best) { best = v; p = r; }
    }
    piv[c] = p;
    if (p != c)
      for (int q = 0; q < 125; ++q) std::swap(A[c * 125 + q], A[p * 125 + q]);
    double dinv = 1.0 / A[c * 125 + c];
    for (int r = c + 1; r < 125; ++r) {
      double f = A[r * 125 + c] * dinv;
      A[r * 125 + c] = f;
      if (f != 0.0)
        for (int q = c + 1; q < 125; ++q) A[r * 125 + q] -= f * A[c * 125 + q];
    }
  }
  double x[125];
  for (int i = 0; i < 125; ++i) x[i] = std::sin(0.7 * i + 0.3);
  for (int it = 0; it < 4; ++it) {
    double y[125];
    for (int i = 0; i < 125; ++i) y[i] = x[i];
    for (int c = 0; c < 125; ++c)
      if (piv[c] != c) std::swap(y[c], y[piv[c]]);
    for (int r = 0; r < 125; ++r) {
      double acc = y[r];
      for (int q = 0; q < r; ++q) acc -= A[r * 125 + q] * y[q];
      y[r] = acc;
    }
    for (int r = 124; r >= 0; --r) {
      double acc = y[r];
      for (int q = r + 1; q < 125; ++q) acc -= A[r * 125 + q] * y[q];
      y[r] = acc / A[r * 125 + r];
    }
    double n = 0.0;
    for (int i = 0; i < 125; ++i) n += y[i] * y[i];
    n = std::sqrt(n);
    for (int i = 0; i < 125; ++i) x[i] = y[i] / n;
  }
  int am = 0;
  for (int i = 1; i < 125; ++i)
    if (std::fabs(x[i]) > std::fabs(x[am])) am = i;
  double sg = (x[am] < 0.0) ? -1.0 : 1.0;
  for (int i = 0; i < 125; ++i) outc[i] = (float)(sg * x[i]);
}

extern "C" void kernel_launch(void* const* d_in, const int* in_sizes, int n_in,
                              void* d_out, int out_size, void* d_ws, size_t ws_size,
                              hipStream_t stream) {
  (void)n_in; (void)d_ws; (void)ws_size; (void)out_size;
  C222Arg cc;
  compute_c222(cc.c);
  const float* g_s    = (const float*)d_in[0];
  const float* g_t2s  = (const float*)d_in[1];
  const float* g_w000 = (const float*)d_in[2];
  const float* g_w110 = (const float*)d_in[3];
  const float* g_w011 = (const float*)d_in[4];
  const float* g_w101 = (const float*)d_in[5];
  const float* g_w111 = (const float*)d_in[6];
  const float* g_v010 = (const float*)d_in[7];
  const float* g_v100 = (const float*)d_in[8];
  const float* g_v110 = (const float*)d_in[9];
  int Btot = in_sizes[0] / 16;
  int blocks = (Btot + WAVES * 16 - 1) / (WAVES * 16);
  eqhead_kernel<<<blocks, THREADS, 0, stream>>>(g_s, g_t2s, g_w000, g_w110, g_w011,
                                                g_w101, g_w111, g_v010, g_v100,
                                                g_v110, (float*)d_out, Btot, cc);
}
